// LowRankAttention_88175678587239
// MI455X (gfx1250) — matrix operand order repaired and checked by reading, not compiled, not run on hardware
//
#include <hip/hip_runtime.h>

// ---------------------------------------------------------------------------
// LowRankAttention for MI455X (gfx1250, wave32, WMMA + TDM)
// B=4,H=16 -> 64 independent heads. S=4096, D=R=64.
// Big contractions: v_wmma_f32_16x16x32_bf16 (f32 accumulate).
// Small per-head matrices + rp column blocks staged into LDS via the
// Tensor Data Mover (tensor_load_to_lds, TENSORcnt), falling back to
// cooperative copies if the builtin is unavailable.
// ---------------------------------------------------------------------------

#define S_DIM 4096
#define D_DIM 64
#define R_DIM 64
#define BH    64
#define NUM_STEPS 10
#define STEP_SIZE 1.0f
#define PERTURB 1.0e-3f
#define INV_S2 (1.0f / ((float)S_DIM * (float)S_DIM))
#define GRAM_SPLIT 4
#define GRAM_CHUNK (S_DIM / GRAM_SPLIT)

typedef __attribute__((ext_vector_type(16))) __bf16 v16bf;
typedef __attribute__((ext_vector_type(8)))  float  v8f;
typedef __attribute__((ext_vector_type(4)))  unsigned v4u;
typedef __attribute__((ext_vector_type(4)))  int      v4i;
typedef __attribute__((ext_vector_type(8)))  int      v8i;

#if __has_builtin(__builtin_amdgcn_tensor_load_to_lds)
#define USE_TDM 1
#else
#define USE_TDM 0
#endif

__device__ __forceinline__ v8f vzero8() {
  v8f z = {0.f, 0.f, 0.f, 0.f, 0.f, 0.f, 0.f, 0.f};
  return z;
}

__device__ __forceinline__ v8f wmma_bf16(v16bf a, v16bf b, v8f c) {
  return __builtin_amdgcn_wmma_f32_16x16x32_bf16(false, a, false, b, (short)0, c,
                                                 false, false);
}

#if USE_TDM
// Build a D# (ISA 8.3/8.4) for a 2D tile of f32 and issue TENSOR_LOAD_TO_LDS.
// tileX elements per row, tileY rows, row stride strideX (f32 units).
// LDS dest is packed row-major [tileY][tileX].
__device__ __forceinline__ void tdm_load_2d(unsigned ldsOff, const void* gptr,
                                            unsigned tileX, unsigned tileY,
                                            unsigned strideX) {
  unsigned long long ga = (unsigned long long)(size_t)gptr;
  v4u g0;
  g0[0] = 1u;                                     // count=1, user descriptor
  g0[1] = ldsOff;                                 // lds_addr (bytes)
  g0[2] = (unsigned)(ga & 0xFFFFFFFFu);           // global_addr[31:0]
  g0[3] = (unsigned)((ga >> 32) & 0x01FFFFFFu) |  // global_addr[56:32]
          (2u << 30);                             // type=2 ("image")
  v8i g1;
  g1[0] = (int)(2u << 16);                        // data_size=4B, mask/flags=0
  g1[1] = (int)((tileX & 0xFFFFu) << 16);         // tensor_dim0[15:0]
  g1[2] = (int)(((tileX >> 16) & 0xFFFFu) |       // tensor_dim0[31:16]
                ((tileY & 0xFFFFu) << 16));       // tensor_dim1[15:0]
  g1[3] = (int)(((tileY >> 16) & 0xFFFFu) |       // tensor_dim1[31:16]
                ((tileX & 0xFFFFu) << 16));       // tile_dim0
  g1[4] = (int)(tileY & 0xFFFFu);                 // tile_dim1 (tile_dim2=0)
  g1[5] = (int)strideX;                           // tensor_dim0_stride[31:0]
  g1[6] = 0;                                      // stride hi / dim1_stride
  g1[7] = 0;
  v4i z4 = {0, 0, 0, 0};
#if defined(__clang_major__) && (__clang_major__ >= 23)
  v8i z8 = {0, 0, 0, 0, 0, 0, 0, 0};
  __builtin_amdgcn_tensor_load_to_lds(g0, g1, z4, z4, z8, 0);
#else
  __builtin_amdgcn_tensor_load_to_lds(g0, g1, z4, z4, 0);
#endif
}
#endif

// Load two 8-float contiguous blocks (b128 x4), scale-square, pack to bf16.
__device__ __forceinline__ v16bf bf16_sq8x2(const float* p0, const float* p1,
                                            float ir2) {
  float4 a0 = ((const float4*)p0)[0];
  float4 a1 = ((const float4*)p0)[1];
  float4 a2 = ((const float4*)p1)[0];
  float4 a3 = ((const float4*)p1)[1];
  float t[16] = {a0.x, a0.y, a0.z, a0.w, a1.x, a1.y, a1.z, a1.w,
                 a2.x, a2.y, a2.z, a2.w, a3.x, a3.y, a3.z, a3.w};
  v16bf r;
#pragma unroll
  for (int e = 0; e < 16; ++e) {
    float u = t[e];
    r[e] = (__bf16)(u * u * ir2);
  }
  return r;
}
__device__ __forceinline__ v16bf bf16_sq16(const float* p, float ir2) {
  return bf16_sq8x2(p, p + 8, ir2);
}

// ------------------------- init (hash RNG) ---------------------------------
__device__ __forceinline__ unsigned hash_u32(unsigned x) {
  x ^= x >> 16; x *= 0x7feb352dU; x ^= x >> 15; x *= 0x846ca68bU; x ^= x >> 16;
  return x;
}
__device__ __forceinline__ float unif_pm(unsigned seed) {
  unsigned h = hash_u32(seed);
  float u = (float)(h >> 8) * (1.0f / 16777216.0f);
  return 2.0f * PERTURB * u - PERTURB;
}

__global__ __launch_bounds__(256) void k_init(float* lp, float* rp) {
  size_t i = (size_t)blockIdx.x * blockDim.x + threadIdx.x;
  const size_t N = (size_t)BH * S_DIM * R_DIM;
  if (i < N) {
    lp[i] = 0.125f    + unif_pm((unsigned)i * 2654435761u ^ 0x9e3779b9u);
    rp[i] = 0.015625f + unif_pm((unsigned)i * 2246822519u ^ 0x85ebca6bu);
  }
}

__global__ __launch_bounds__(256) void k_zero(float* p, int n) {
  int i = blockIdx.x * 256 + threadIdx.x;
  if (i < n) p[i] = 0.f;
}

// ------------------------- row norms ---------------------------------------
__global__ __launch_bounds__(256) void k_norm_l(const float* __restrict__ lp,
                                                float* __restrict__ iln) {
  int wave = threadIdx.x >> 5, lane = threadIdx.x & 31;
  size_t row = (size_t)blockIdx.x * 8 + wave; // BH*S rows total
  const float2* p = (const float2*)(lp + row * R_DIM);
  float2 x = p[lane];
  float s = x.x * x.x + x.y * x.y;
#pragma unroll
  for (int m = 1; m < 32; m <<= 1) s += __shfl_xor(s, m, 32);
  if (lane == 0) iln[row] = rsqrtf(s);
}

__global__ __launch_bounds__(256) void k_norm_r(const float* __restrict__ rp,
                                                float* __restrict__ irn,
                                                float* __restrict__ dotb) {
  int row = blockIdx.x; // BH*R rows
  const float4* p = (const float4*)(rp + (size_t)row * S_DIM);
  float s = 0.f;
  for (int i = threadIdx.x; i < S_DIM / 4; i += 256) {
    float4 x = p[i];
    s += x.x * x.x + x.y * x.y + x.z * x.z + x.w * x.w;
  }
  __shared__ float red[256];
  red[threadIdx.x] = s;
  __syncthreads();
  for (int off = 128; off > 0; off >>= 1) {
    if (threadIdx.x < (unsigned)off) red[threadIdx.x] += red[threadIdx.x + off];
    __syncthreads();
  }
  if (threadIdx.x == 0) {
    irn[row] = rsqrtf(red[0]);
    dotb[row] = 0.f;
  }
}

// ------------------------- Gram reductions over S --------------------------
// MODE 0: M1[r,l]   = sum_s rsq(r,s) rsq(l,s)
// MODE 1: LtL[k,l]  = sum_s lsq(s,k) lsq(s,l)
// MODE 2: KtRt[a,l] = sum_s key[s,a] rsq(l,s)
// MODE 3: LtQ[k,a]  = sum_s lsq(s,k) query[s,a]
// MODE 4: RV[r,a]   = sum_s rsq(r,s) value[s,a]
template <int MODE>
__global__ __launch_bounds__(256) void k_gram_t(
    const float* __restrict__ q, const float* __restrict__ key,
    const float* __restrict__ val, const float* __restrict__ lp,
    const float* __restrict__ rp, const float* __restrict__ iln,
    const float* __restrict__ irn, float* __restrict__ outg) {
  const int head = blockIdx.y;
  const float* qh = q + (size_t)head * S_DIM * D_DIM;
  const float* kh = key + (size_t)head * S_DIM * D_DIM;
  const float* vh = val + (size_t)head * S_DIM * D_DIM;
  const float* lph = lp + (size_t)head * S_DIM * R_DIM;
  const float* rph = rp + (size_t)head * R_DIM * S_DIM;
  const float* ilnh = iln + (size_t)head * S_DIM;
  const float* irnh = irn + (size_t)head * R_DIM;
  float* outp = outg + (size_t)head * 64 * 64;
  (void)qh; (void)kh; (void)vh;

  const int wave = threadIdx.x >> 5;
  const int lane = threadIdx.x & 31;
  const int half = lane >> 4, ln15 = lane & 15;
  const int t0 = wave * 2; // two adjacent n-tiles share the A tile
  const int mt = t0 >> 2;
  const int nt0 = t0 & 3, nt1 = nt0 + 1;
  const int ra = mt * 16 + ln15;
  const int n0 = nt0 * 16 + ln15, n1 = nt1 * 16 + ln15;

  float ir2a = 0.f, ir2b0 = 0.f, ir2b1 = 0.f;
  if constexpr (MODE == 0 || MODE == 4) ir2a = irnh[ra] * irnh[ra];
  if constexpr (MODE == 0 || MODE == 2) {
    ir2b0 = irnh[n0] * irnh[n0];
    ir2b1 = irnh[n1] * irnh[n1];
  }

  v8f acc0 = vzero8(), acc1 = vzero8();
  const int sBeg = blockIdx.x * GRAM_CHUNK;
  const int sEnd = sBeg + GRAM_CHUNK;
  for (int s0 = sBeg; s0 < sEnd; s0 += 32) {
    v16bf a, b0, b1;
    // ---- A operand: rows m, K pattern: half*16+e (16 contiguous in s)
    if constexpr (MODE == 0 || MODE == 4) {
      const float* base = rph + (size_t)ra * S_DIM + s0;
      __builtin_prefetch(base + 2048, 0, 3);
      a = bf16_sq8x2(base + half * 16, base + half * 16 + 8, ir2a);
    } else if constexpr (MODE == 2) {
#pragma unroll
      for (int e = 0; e < 16; ++e) {
        int kk = half * 16 + e;
        a[e] = (__bf16)kh[(size_t)(s0 + kk) * D_DIM + ra];
      }
    } else { // 1, 3: column of lp, normalized-squared
      __builtin_prefetch(&lph[(size_t)(s0 + 512) * R_DIM + ra], 0, 3);
#pragma unroll
      for (int e = 0; e < 16; ++e) {
        int kk = half * 16 + e;
        float t = lph[(size_t)(s0 + kk) * R_DIM + ra] * ilnh[s0 + kk];
        a[e] = (__bf16)(t * t);
      }
    }
    // ---- B operand: cols n, K pattern: half*16+e (16 contiguous in s)
    if constexpr (MODE == 0 || MODE == 2) {
      b0 = bf16_sq16(rph + (size_t)n0 * S_DIM + s0 + half * 16, ir2b0);
      b1 = bf16_sq16(rph + (size_t)n1 * S_DIM + s0 + half * 16, ir2b1);
    } else if constexpr (MODE == 1) {
#pragma unroll
      for (int e = 0; e < 16; ++e) {
        int kk = half * 16 + e;
        float il = ilnh[s0 + kk];
        float u0 = lph[(size_t)(s0 + kk) * R_DIM + n0] * il;
        float u1 = lph[(size_t)(s0 + kk) * R_DIM + n1] * il;
        b0[e] = (__bf16)(u0 * u0);
        b1[e] = (__bf16)(u1 * u1);
      }
    } else if constexpr (MODE == 3) {
#pragma unroll
      for (int e = 0; e < 16; ++e) {
        int kk = half * 16 + e;
        b0[e] = (__bf16)qh[(size_t)(s0 + kk) * D_DIM + n0];
        b1[e] = (__bf16)qh[(size_t)(s0 + kk) * D_DIM + n1];
      }
    } else { // 4: value (streamed once -> non-temporal)
#pragma unroll
      for (int e = 0; e < 16; ++e) {
        int kk = half * 16 + e;
        b0[e] = (__bf16)__builtin_nontemporal_load(&vh[(size_t)(s0 + kk) * D_DIM + n0]);
        b1[e] = (__bf16)__builtin_nontemporal_load(&vh[(size_t)(s0 + kk) * D_DIM + n1]);
      }
    }
    acc0 = wmma_bf16(a, b0, acc0);
    acc1 = wmma_bf16(a, b1, acc1);
  }
#pragma unroll
  for (int vv = 0; vv < 8; ++vv) {
    int rowOut = mt * 16 + vv + 8 * half;
    atomicAdd(&outp[rowOut * 64 + nt0 * 16 + ln15], acc0[vv]);
    atomicAdd(&outp[rowOut * 64 + nt1 * 16 + ln15], acc1[vv]);
  }
}

// ------------------------- right-factor update -----------------------------
// dRraw = (LtL@right - LtQ@keyT) * inv_s2 ; dR = 2*rs*dRraw
// pass 0: dot[r] += <rs_row, dR_row> (atomics). pass 1: projected update.
// LtL/LtQ and the 64x32 rp column block are DMA'd into LDS by the TDM.
__global__ __launch_bounds__(256) void k_update_right(
    const float* __restrict__ key, float* __restrict__ rp,
    const float* __restrict__ irn, const float* __restrict__ LtL,
    const float* __restrict__ LtQ, float* __restrict__ dotb, int pass) {
  const int head = blockIdx.y;
  const float* kh = key + (size_t)head * S_DIM * D_DIM;
  float* rph = rp + (size_t)head * R_DIM * S_DIM;
  const float* irnh = irn + (size_t)head * R_DIM;
  float* doth = dotb + (size_t)head * R_DIM;
  const int cBase = blockIdx.x * 32;

  __shared__ float sL[64 * 64];
  __shared__ float sQ[64 * 64];
  __shared__ float sRp[64 * 32]; // rp rows 0..63, cols cBase..cBase+31
  const float* Lh = LtL + (size_t)head * 64 * 64;
  const float* Qh = LtQ + (size_t)head * 64 * 64;
#if USE_TDM
  if (threadIdx.x == 0) {
    tdm_load_2d((unsigned)(size_t)sL, Lh, 4096, 1, 4096);
    tdm_load_2d((unsigned)(size_t)sQ, Qh, 4096, 1, 4096);
    tdm_load_2d((unsigned)(size_t)sRp, rph + cBase, 32, 64, S_DIM);
    __builtin_amdgcn_s_wait_tensorcnt(0);
  }
#else
  for (int i = threadIdx.x; i < 64 * 64; i += 256) {
    sL[i] = Lh[i];
    sQ[i] = Qh[i];
  }
  for (int i = threadIdx.x; i < 64 * 32; i += 256)
    sRp[i] = rph[(size_t)(i >> 5) * S_DIM + cBase + (i & 31)];
#endif
  __syncthreads();

  const int wave = threadIdx.x >> 5;
  const int lane = threadIdx.x & 31;
  const int half = lane >> 4, ln15 = lane & 15;
  const int rt = wave & 3;             // r-tile 0..3
  const int lc0 = (wave >> 2) * 16;    // local column base (0 or 16)

  v8f acc3 = vzero8(), acc4 = vzero8();
#pragma unroll
  for (int kc = 0; kc < 2; ++kc) {
    const int s0 = kc * 32;
    v16bf a3, a4, b3, b4;
#pragma unroll
    for (int e = 0; e < 16; ++e) {
      int kkA = half * 16 + e;
      a3[e] = (__bf16)sL[(rt * 16 + ln15) * 64 + s0 + kkA];
      a4[e] = (__bf16)sQ[(rt * 16 + ln15) * 64 + s0 + kkA];
    }
#pragma unroll
    for (int e = 0; e < 16; ++e) {
      int kkB = half * 16 + e;
      float rv = sRp[(s0 + kkB) * 32 + lc0 + ln15] * irnh[s0 + kkB];
      b3[e] = (__bf16)(rv * rv);
      b4[e] = (__bf16)kh[(size_t)(cBase + lc0 + ln15) * D_DIM + s0 + kkB];
    }
    acc3 = wmma_bf16(a3, b3, acc3);
    acc4 = wmma_bf16(a4, b4, acc4);
  }

#pragma unroll
  for (int vv = 0; vv < 8; ++vv) {
    int r = rt * 16 + vv + 8 * half;
    int lc = lc0 + ln15;
    float rsv = sRp[r * 32 + lc] * irnh[r];
    float draw = (acc3[vv] - acc4[vv]) * INV_S2;
    float dR = 2.f * rsv * draw;
    if (pass == 0) {
      float p = rsv * dR;
#pragma unroll
      for (int m = 1; m < 16; m <<= 1) p += __shfl_xor(p, m, 16);
      if (ln15 == 0) atomicAdd(&doth[r], p);
    } else {
      float d = (dR - rsv * doth[r]) * irnh[r];
      rph[(size_t)r * S_DIM + cBase + lc] -= STEP_SIZE * d;
    }
  }
}

// ------------------------- left-factor update (fully fused) ----------------
// T1 = left@M1, T2 = query@KtRt; dL = (T1-T2)*inv_s2*2*ls; project; lp -= dL/||lp||
__global__ __launch_bounds__(256) void k_update_left(
    const float* __restrict__ q, float* __restrict__ lp,
    const float* __restrict__ iln, const float* __restrict__ M1,
    const float* __restrict__ KtRt) {
  const int head = blockIdx.y;
  const float* qh = q + (size_t)head * S_DIM * D_DIM;
  float* lph = lp + (size_t)head * S_DIM * R_DIM;
  const float* ilnh = iln + (size_t)head * S_DIM;

  __shared__ float sM[64 * 64];
  __shared__ float sK[64 * 64];
  const float* Mh = M1 + (size_t)head * 64 * 64;
  const float* Kh = KtRt + (size_t)head * 64 * 64;
#if USE_TDM
  if (threadIdx.x == 0) {
    tdm_load_2d((unsigned)(size_t)sM, Mh, 4096, 1, 4096);
    tdm_load_2d((unsigned)(size_t)sK, Kh, 4096, 1, 4096);
    __builtin_amdgcn_s_wait_tensorcnt(0);
  }
#else
  for (int i = threadIdx.x; i < 64 * 64; i += 256) {
    sM[i] = Mh[i];
    sK[i] = Kh[i];
  }
#endif
  __syncthreads();

  const int wave = threadIdx.x >> 5;
  const int lane = threadIdx.x & 31;
  const int half = lane >> 4, ln15 = lane & 15;
  const int j0 = blockIdx.x * 128 + wave * 16; // 16 rows per wave

  v8f t1[4], t2[4];
#pragma unroll
  for (int ct = 0; ct < 4; ++ct) { t1[ct] = vzero8(); t2[ct] = vzero8(); }

#pragma unroll
  for (int kc = 0; kc < 2; ++kc) {
    const int s0 = kc * 32;
    v16bf a1, a2;
    const int rowA = j0 + ln15;
    const float ila = ilnh[rowA];
#pragma unroll
    for (int e = 0; e < 16; ++e) {
      int kkA = half * 16 + e;
      float t = lph[(size_t)rowA * R_DIM + s0 + kkA] * ila;
      a1[e] = (__bf16)(t * t);
      a2[e] = (__bf16)qh[(size_t)rowA * D_DIM + s0 + kkA];
    }
#pragma unroll
    for (int ct = 0; ct < 4; ++ct) {
      v16bf b1, b2;
#pragma unroll
      for (int e = 0; e < 16; ++e) {
        int kkB = half * 16 + e;
        b1[e] = (__bf16)sM[(s0 + kkB) * 64 + ct * 16 + ln15];
        b2[e] = (__bf16)sK[(s0 + kkB) * 64 + ct * 16 + ln15];
      }
      t1[ct] = wmma_bf16(a1, b1, t1[ct]);
      t2[ct] = wmma_bf16(a2, b2, t2[ct]);
    }
  }

  float lsv[4][8], dlv[4][8], dots[8];
#pragma unroll
  for (int vv = 0; vv < 8; ++vv) {
    int j = j0 + vv + 8 * half;
    float il = ilnh[j];
    float p = 0.f;
#pragma unroll
    for (int ct = 0; ct < 4; ++ct) {
      int col = ct * 16 + ln15;
      float ls = lph[(size_t)j * R_DIM + col] * il;
      float d = (t1[ct][vv] - t2[ct][vv]) * INV_S2;
      float dl = d * 2.f * ls;
      lsv[ct][vv] = ls;
      dlv[ct][vv] = dl;
      p += ls * dl;
    }
#pragma unroll
    for (int m = 1; m < 16; m <<= 1) p += __shfl_xor(p, m, 16);
    dots[vv] = p; // per-row <ls, dL>, replicated across the half-wave
  }
#pragma unroll
  for (int vv = 0; vv < 8; ++vv) {
    int j = j0 + vv + 8 * half;
    float il = ilnh[j];
#pragma unroll
    for (int ct = 0; ct < 4; ++ct) {
      int col = ct * 16 + ln15;
      float dl = (dlv[ct][vv] - lsv[ct][vv] * dots[vv]) * il;
      lph[(size_t)j * R_DIM + col] -= STEP_SIZE * dl;
    }
  }
}

// ------------------------- final: out = left @ RV --------------------------
__global__ __launch_bounds__(256) void k_final_out(float* __restrict__ out,
                                                   const float* __restrict__ lp,
                                                   const float* __restrict__ iln,
                                                   const float* __restrict__ RV) {
  const int head = blockIdx.y;
  float* outh = out + (size_t)head * S_DIM * D_DIM;
  const float* lph = lp + (size_t)head * S_DIM * R_DIM;
  const float* ilnh = iln + (size_t)head * S_DIM;

  __shared__ float sR[64 * 64];
  const float* Rh = RV + (size_t)head * 64 * 64;
#if USE_TDM
  if (threadIdx.x == 0) {
    tdm_load_2d((unsigned)(size_t)sR, Rh, 4096, 1, 4096);
    __builtin_amdgcn_s_wait_tensorcnt(0);
  }
#else
  for (int i = threadIdx.x; i < 64 * 64; i += 256) sR[i] = Rh[i];
#endif
  __syncthreads();

  const int wave = threadIdx.x >> 5;
  const int lane = threadIdx.x & 31;
  const int half = lane >> 4, ln15 = lane & 15;
  const int j0 = blockIdx.x * 128 + wave * 16;

  v8f acc[4];
#pragma unroll
  for (int ct = 0; ct < 4; ++ct) acc[ct] = vzero8();

#pragma unroll
  for (int kc = 0; kc < 2; ++kc) {
    const int s0 = kc * 32;
    v16bf a;
    const int rowA = j0 + ln15;
    const float ila = ilnh[rowA];
#pragma unroll
    for (int e = 0; e < 16; ++e) {
      int kkA = half * 16 + e;
      float t = lph[(size_t)rowA * R_DIM + s0 + kkA] * ila;
      a[e] = (__bf16)(t * t);
    }
#pragma unroll
    for (int ct = 0; ct < 4; ++ct) {
      v16bf b;
#pragma unroll
      for (int e = 0; e < 16; ++e) {
        int kkB = half * 16 + e;
        b[e] = (__bf16)sR[(s0 + kkB) * 64 + ct * 16 + ln15];
      }
      acc[ct] = wmma_bf16(a, b, acc[ct]);
    }
  }
#pragma unroll
  for (int vv = 0; vv < 8; ++vv) {
    int j = j0 + vv + 8 * half;
#pragma unroll
    for (int ct = 0; ct < 4; ++ct)
      outh[(size_t)j * D_DIM + ct * 16 + ln15] = acc[ct][vv];
  }
}

// ---------------------------------------------------------------------------
extern "C" void kernel_launch(void* const* d_in, const int* in_sizes, int n_in,
                              void* d_out, int out_size, void* d_ws,
                              size_t ws_size, hipStream_t stream) {
  (void)in_sizes; (void)n_in; (void)out_size; (void)ws_size;
  const float* q = (const float*)d_in[0];
  const float* key = (const float*)d_in[1];
  const float* val = (const float*)d_in[2];
  float* out = (float*)d_out;

  const size_t NLR = (size_t)BH * S_DIM * R_DIM; // 16,777,216 floats
  const size_t NMAT = (size_t)BH * 64 * 64;      // 262,144 floats
  float* ws = (float*)d_ws;
  float* lp = ws;   ws += NLR;
  float* rp = ws;   ws += NLR;
  float* iln = ws;  ws += (size_t)BH * S_DIM;
  float* irn = ws;  ws += (size_t)BH * R_DIM;
  float* dotb = ws; ws += (size_t)BH * R_DIM;
  float* M1 = ws;   ws += NMAT;
  float* LtL = ws;  ws += NMAT;
  float* KtRt = ws; ws += NMAT;
  float* LtQ = ws;  ws += NMAT;
  float* RV = ws;   ws += NMAT;

  {
    int blocks = (int)((NLR + 255) / 256);
    k_init<<<blocks, 256, 0, stream>>>(lp, rp);
  }

  const dim3 gGram(GRAM_SPLIT, BH);
  for (int step = 0; step < NUM_STEPS; ++step) {
    k_norm_l<<<(BH * S_DIM) / 8, 256, 0, stream>>>(lp, iln);
    k_norm_r<<<BH * R_DIM, 256, 0, stream>>>(rp, irn, dotb);
    // zero M1..LtQ (contiguous 4*NMAT region starting at M1)
    k_zero<<<(int)((4 * NMAT + 255) / 256), 256, 0, stream>>>(M1, (int)(4 * NMAT));
    k_gram_t<0><<<gGram, 256, 0, stream>>>(q, key, val, lp, rp, iln, irn, M1);
    k_gram_t<1><<<gGram, 256, 0, stream>>>(q, key, val, lp, rp, iln, irn, LtL);
    k_gram_t<2><<<gGram, 256, 0, stream>>>(q, key, val, lp, rp, iln, irn, KtRt);
    k_gram_t<3><<<gGram, 256, 0, stream>>>(q, key, val, lp, rp, iln, irn, LtQ);
    k_update_right<<<dim3(S_DIM / 32, BH), 256, 0, stream>>>(key, rp, irn, LtL,
                                                             LtQ, dotb, 0);
    k_update_right<<<dim3(S_DIM / 32, BH), 256, 0, stream>>>(key, rp, irn, LtL,
                                                             LtQ, dotb, 1);
    k_update_left<<<dim3(S_DIM / 128, BH), 256, 0, stream>>>(q, lp, iln, M1,
                                                             KtRt);
  }

  // final: norms of converged factors, RV = right@value, out = left@RV
  k_norm_l<<<(BH * S_DIM) / 8, 256, 0, stream>>>(lp, iln);
  k_norm_r<<<BH * R_DIM, 256, 0, stream>>>(rp, irn, dotb);
  k_zero<<<(int)((NMAT + 255) / 256), 256, 0, stream>>>(RV, (int)NMAT);
  k_gram_t<4><<<gGram, 256, 0, stream>>>(q, key, val, lp, rp, iln, irn, RV);
  k_final_out<<<dim3(S_DIM / 128, BH), 256, 0, stream>>>(out, lp, iln, RV);
}
